// LSTMModel_781684048107
// MI455X (gfx1250) — matrix operand order repaired and checked by reading, not compiled, not run on hardware
//
#include <hip/hip_runtime.h>
#include <hip/hip_bf16.h>

typedef __attribute__((ext_vector_type(16))) __bf16 v16bf;
typedef __attribute__((ext_vector_type(8)))  float  v8f;

#define B_    64
#define S_    2048
#define E_    128
#define H_    256
#define O_    16
#define FOURH 1024

// ---- d_ws layout (__bf16 element offsets) ----
#define W0F_OFF 0
#define U0F_OFF (128 * 1024)
#define W1F_OFF (128 * 1024 + 256 * 1024)
#define U1F_OFF (128 * 1024 + 2 * 256 * 1024)
// total ws need: (128+3*256)*1024 * 2 bytes = 1,835,008 B

// ---- dynamic LDS layout (byte offsets) ----
#define G_OFF    0                         // float[16][1024]  gate pre-activations
#define C0_OFF   (16 * 1024 * 4)           // float[16][256]
#define C1_OFF   (C0_OFF + 16 * 256 * 4)   // float[16][256]
#define XA_OFF   (C1_OFF + 16 * 256 * 4)   // __bf16 [4 kt][32 lane][16 e]
#define H0A_OFF  (XA_OFF + 4 * 512 * 2)    // __bf16 [8 kt][32][16]
#define H1A_OFF  (H0A_OFF + 8 * 512 * 2)   // __bf16 [8 kt][32][16]
#define SMEM_BYTES (H1A_OFF + 8 * 512 * 2) // 118,784 bytes

__device__ __forceinline__ float sig_(float x) { return 1.0f / (1.0f + __expf(-x)); }

// B-fragment (32x16, K x N) lane-major packing of a row-major [K,1024] fp32 weight:
// dst[(kt*64+nt)*512 + lane*16 + e] = (bf16) W[(kt*32 + (e&7) + 8*(lane>>4) + 16*(e>>3)) * 1024 + nt*16 + (lane&15)]
__global__ void pack_w_kernel(const float* __restrict__ src, __bf16* __restrict__ dst, int total) {
    int idx = blockIdx.x * blockDim.x + threadIdx.x;
    if (idx >= total) return;
    int e    = idx & 15;
    int lane = (idx >> 4) & 31;
    int tile = idx >> 9;              // kt*64 + nt
    int nt   = tile & 63;
    int kt   = tile >> 6;
    int n    = nt * 16 + (lane & 15);
    int half = lane >> 4;
    int k    = kt * 32 + (e & 7) + 8 * half + 16 * (e >> 3);
    dst[idx] = (__bf16)src[k * FOURH + n];
}

// One wave computes a 16 x 64 slice (4 N-tiles) of g = A1*B1 + A2*B2 + bias.
// A fragments come from LDS (lane-major 16x32 tiles), B fragments from packed global.
__device__ __forceinline__ void gemm_slice(const __bf16* __restrict__ aF1, int nkt1,
                                           const __bf16* __restrict__ bW1,
                                           const __bf16* __restrict__ aF2, int nkt2,
                                           const __bf16* __restrict__ bW2,
                                           const float* __restrict__ bias,
                                           float* __restrict__ g,
                                           int wave, int lane) {
    const int nt0  = wave * 4;
    const int ncol = lane & 15;
    const int half = lane >> 4;
    v8f acc[4];
#pragma unroll
    for (int t = 0; t < 4; ++t) {
        float bv = bias[(nt0 + t) * 16 + ncol];
        v8f a;
#pragma unroll
        for (int r = 0; r < 8; ++r) a[r] = bv;
        acc[t] = a;
    }
#pragma unroll 2
    for (int kt = 0; kt < nkt1; ++kt) {
        v16bf a = *(const v16bf*)(aF1 + kt * 512 + lane * 16);
#pragma unroll
        for (int t = 0; t < 4; ++t) {
            v16bf b = *(const v16bf*)(bW1 + (kt * 64 + nt0 + t) * 512 + lane * 16);
            acc[t] = __builtin_amdgcn_wmma_f32_16x16x32_bf16(false, a, false, b,
                                                             (short)0, acc[t], false, false);
        }
    }
#pragma unroll 2
    for (int kt = 0; kt < nkt2; ++kt) {
        v16bf a = *(const v16bf*)(aF2 + kt * 512 + lane * 16);
#pragma unroll
        for (int t = 0; t < 4; ++t) {
            v16bf b = *(const v16bf*)(bW2 + (kt * 64 + nt0 + t) * 512 + lane * 16);
            acc[t] = __builtin_amdgcn_wmma_f32_16x16x32_bf16(false, a, false, b,
                                                             (short)0, acc[t], false, false);
        }
    }
    // C/D layout: VGPR r, lane -> row = r + 8*(lane>>4), col = nt*16 + (lane&15)
#pragma unroll
    for (int t = 0; t < 4; ++t)
#pragma unroll
        for (int r = 0; r < 8; ++r)
            g[(r + 8 * half) * FOURH + (nt0 + t) * 16 + ncol] = acc[t][r];
}

// Elementwise LSTM cell update; writes new h into an A-fragment-order bf16 LDS buffer.
__device__ __forceinline__ void cell_update(const float* __restrict__ g,
                                            float* __restrict__ c,
                                            __bf16* __restrict__ hA,
                                            int tid, int nthreads) {
    for (int idx = tid; idx < 16 * H_; idx += nthreads) {
        int m = idx >> 8, h = idx & (H_ - 1);
        float it = sig_(g[m * FOURH + h]);
        float ft = sig_(g[m * FOURH + H_ + h]);
        float ot = sig_(g[m * FOURH + 2 * H_ + h]);
        float ch = tanhf(g[m * FOURH + 3 * H_ + h]);
        float cv = ft * c[idx] + it * ch;
        c[idx]   = cv;
        float hv = ot * tanhf(cv);
        // inverse A-fragment mapping: k' = h&31 -> (half, e)
        int kp  = h & 31;
        int kt  = h >> 5;
        int hlf = (kp >> 3) & 1;
        int e   = (kp & 7) | ((kp >> 4) << 3);
        hA[kt * 512 + (hlf * 16 + m) * 16 + e] = (__bf16)hv;
    }
}

__global__ void __launch_bounds__(512)
lstm_persistent_kernel(const int* __restrict__ tokens, const float* __restrict__ emb,
                       const float* __restrict__ b0, const float* __restrict__ b1,
                       const float* __restrict__ Wout, const float* __restrict__ bout,
                       const __bf16* __restrict__ w0f, const __bf16* __restrict__ u0f,
                       const __bf16* __restrict__ w1f, const __bf16* __restrict__ u1f,
                       float* __restrict__ out) {
    extern __shared__ __align__(32) char smem[];
    float*  g   = (float*)(smem + G_OFF);
    float*  c0  = (float*)(smem + C0_OFF);
    float*  c1  = (float*)(smem + C1_OFF);
    __bf16* xA  = (__bf16*)(smem + XA_OFF);
    __bf16* h0A = (__bf16*)(smem + H0A_OFF);
    __bf16* h1A = (__bf16*)(smem + H1A_OFF);

    const int btile = blockIdx.x;        // 4 blocks x 16 batch rows
    const int tid   = threadIdx.x;
    const int wave  = tid >> 5;
    const int lane  = tid & 31;
    const int NT    = blockDim.x;        // 512

    // zero-init h, c
    for (int idx = tid; idx < 16 * H_; idx += NT) {
        h0A[idx] = (__bf16)0.0f;
        h1A[idx] = (__bf16)0.0f;
        c0[idx]  = 0.0f;
        c1[idx]  = 0.0f;
    }
    __syncthreads();

#pragma unroll 1
    for (int s = 0; s < S_; ++s) {
        // --- gather x_t into A-fragment order (bf16) ---
        for (int idx = tid; idx < 4 * 512; idx += NT) {
            int e     = idx & 15;
            int lane_ = (idx >> 4) & 31;
            int kt    = idx >> 9;
            int m     = lane_ & 15;
            int hlf   = lane_ >> 4;
            int j     = e >> 1, p = e & 1;
            int k     = (j < 4 ? 2 * j + 8 * hlf : 16 + 2 * (j - 4) + 8 * hlf) + p;
            int tok   = tokens[(btile * 16 + m) * S_ + s];
            xA[idx]   = (__bf16)emb[tok * E_ + kt * 32 + k];
        }
        __syncthreads();

        // --- layer 0: g = x@W0 + h0@U0 + b0 ---
        gemm_slice(xA, 4, w0f, h0A, 8, u0f, b0, g, wave, lane);
        __syncthreads();
        cell_update(g, c0, h0A, tid, NT);
        __syncthreads();

        // --- layer 1: g = h0@W1 + h1@U1 + b1 ---
        gemm_slice(h0A, 8, w1f, h1A, 8, u1f, b1, g, wave, lane);
        __syncthreads();
        cell_update(g, c1, h1A, tid, NT);
        __syncthreads();
    }

    // --- logits = h1_last @ Wout + bout  (16x256 @ 256x16, tiny: scalar) ---
    for (int idx = tid; idx < 16 * O_; idx += NT) {
        int m = idx >> 4, o = idx & 15;
        float acc = bout[o];
#pragma unroll 4
        for (int k = 0; k < H_; ++k) {
            int kp = k & 31, kt = k >> 5;
            int hlf = (kp >> 3) & 1;
            int e   = (kp & 7) | ((kp >> 4) << 3);
            float hv = (float)h1A[kt * 512 + (hlf * 16 + m) * 16 + e];
            acc += hv * Wout[k * O_ + o];
        }
        out[(btile * 16 + m) * O_ + o] = acc;
    }

    // --- hn [2,64,256] then cn [2,64,256] ---
    const int HN = B_ * O_;               // 1024: start of hn
    const int CN = HN + 2 * B_ * H_;      // 33792: start of cn
    for (int idx = tid; idx < 16 * H_; idx += NT) {
        int m = idx >> 8, h = idx & (H_ - 1);
        int kp = h & 31, kt = h >> 5;
        int hlf = (kp >> 3) & 1;
        int e   = (kp & 7) | ((kp >> 4) << 3);
        int slot = kt * 512 + (hlf * 16 + m) * 16 + e;
        int gb   = btile * 16 + m;
        out[HN + gb * H_ + h]            = (float)h0A[slot];
        out[HN + B_ * H_ + gb * H_ + h]  = (float)h1A[slot];
        out[CN + gb * H_ + h]            = c0[idx];
        out[CN + B_ * H_ + gb * H_ + h]  = c1[idx];
    }
}

extern "C" void kernel_launch(void* const* d_in, const int* in_sizes, int n_in,
                              void* d_out, int out_size, void* d_ws, size_t ws_size,
                              hipStream_t stream) {
    (void)in_sizes; (void)n_in; (void)out_size; (void)ws_size;
    const int*   tokens = (const int*)d_in[0];
    const float* emb    = (const float*)d_in[1];
    const float* W0     = (const float*)d_in[2];
    const float* U0     = (const float*)d_in[3];
    const float* b0     = (const float*)d_in[4];
    const float* W1     = (const float*)d_in[5];
    const float* U1     = (const float*)d_in[6];
    const float* b1     = (const float*)d_in[7];
    const float* Wout   = (const float*)d_in[8];
    const float* bout   = (const float*)d_in[9];

    __bf16* ws  = (__bf16*)d_ws;
    __bf16* w0f = ws + W0F_OFF;
    __bf16* u0f = ws + U0F_OFF;
    __bf16* w1f = ws + W1F_OFF;
    __bf16* u1f = ws + U1F_OFF;

    // one-time (per launch) weight packing into bf16 B-fragments (L2-resident afterwards)
    {
        int t0 = 128 * 1024, t1 = 256 * 1024;
        pack_w_kernel<<<(t0 + 255) / 256, 256, 0, stream>>>(W0, w0f, t0);
        pack_w_kernel<<<(t1 + 255) / 256, 256, 0, stream>>>(U0, u0f, t1);
        pack_w_kernel<<<(t1 + 255) / 256, 256, 0, stream>>>(W1, w1f, t1);
        pack_w_kernel<<<(t1 + 255) / 256, 256, 0, stream>>>(U1, u1f, t1);
    }

    lstm_persistent_kernel<<<4, 512, SMEM_BYTES, stream>>>(
        tokens, emb, b0, b1, Wout, bout, w0f, u0f, w1f, u1f, (float*)d_out);
}
